// HierarchicalCompressor_73581379715498
// MI455X (gfx1250) — matrix operand order repaired and checked by reading, not compiled, not run on hardware
//
#include <hip/hip_runtime.h>
#include <hip/hip_bf16.h>

// ---------------------------------------------------------------------------
// HierarchicalCompressor for MI455X (gfx1250): bf16 WMMA GEMM pipeline.
// All heavy matmuls lower to v_wmma_f32_16x16x32_bf16 (f32 accumulate).
// GEMM: 128x128 macro tile, 8 waves, each wave 32x64 (8 WMMAs / K-step).
// Async global->LDS staging (global_load_async_to_lds_b128 + s_wait_asynccnt)
// when the toolchain exposes the gfx1250 builtins; sync b128 path otherwise.
// ---------------------------------------------------------------------------

typedef __attribute__((ext_vector_type(16))) __bf16 v16bf;
typedef __attribute__((ext_vector_type(8)))  float  v8f;

constexpr int Bc = 4, Sc = 2048, Dc = 4096, Tc = 256, Hc = 8, DHc = 512;
constexpr int Wc = 32, NWc = 64;
constexpr int BM = 128, BN = 128, BKc = 32, PADK = 40; // PADK*2B = 80B (16B aligned)

#if defined(__has_builtin)
#if __has_builtin(__builtin_amdgcn_global_load_async_to_lds_b128) && \
    __has_builtin(__builtin_amdgcn_s_wait_asynccnt)
#define HC_ASYNC 1
#endif
#endif
#ifndef HC_ASYNC
#define HC_ASYNC 0
#endif

#if HC_ASYNC
typedef int hc_v4i __attribute__((vector_size(4 * sizeof(int))));
typedef __attribute__((address_space(1))) hc_v4i* hc_gptr;   // global int4*
typedef __attribute__((address_space(3))) hc_v4i* hc_lptr;   // LDS int4*
#endif

__device__ __forceinline__ unsigned short f2bf(float f) {
  unsigned int u = __builtin_bit_cast(unsigned int, f);
  u += 0x7FFFu + ((u >> 16) & 1u);           // round-to-nearest-even
  return (unsigned short)(u >> 16);
}

__device__ __forceinline__ v8f wmma_bf16(v16bf a, v16bf b, v8f c) {
  // 8 args: (neg_a, A, neg_b, B, c_mod, C, reuse_a, reuse_b)
  return __builtin_amdgcn_wmma_f32_16x16x32_bf16(false, a, false, b, (short)0, c,
                                                 false, false);
}

// 16B copy global->LDS; gfx1250 async (ASYNCcnt) path if available.
__device__ __forceinline__ void cp16(const unsigned short* g, unsigned short* l) {
#if HC_ASYNC
  __builtin_amdgcn_global_load_async_to_lds_b128((hc_gptr)g, (hc_lptr)l, 0, 0);
#else
  *reinterpret_cast<uint4*>(l) = *reinterpret_cast<const uint4*>(g);
#endif
}

__device__ __forceinline__ void stage_fence() {
#if HC_ASYNC
  __builtin_amdgcn_s_wait_asynccnt(0);
#endif
  __syncthreads();
}

// A fragment: 16x32 (MxK) bf16, per ISA layout:
// lanes 0-15: M=lane, VGPR0..3 hold K=0..7, VGPR4..7 hold K=16..23
// lanes 16-31: same M, K shifted +8.
__device__ __forceinline__ v16bf ldA16x32(const unsigned short* p, int sm, int sk) {
  int l = threadIdx.x & 31;
  int m = l & 15, g = l >> 4;
  v16bf a;
#pragma unroll
  for (int e = 0; e < 16; ++e) {
    int v = e >> 1, h = e & 1;
    int k = ((v & 4) << 2) + ((v & 3) * 2 + h) + g * 8;
    a[e] = __builtin_bit_cast(__bf16, p[m * sm + k * sk]);
  }
  return a;
}

// B fragment: 32x16 (KxN) bf16: lane = N (two halves), elements walk K.
__device__ __forceinline__ v16bf ldB32x16(const unsigned short* p, int sn, int sk) {
  int l = threadIdx.x & 31;
  int n = l & 15, g = l >> 4;
  v16bf b;
#pragma unroll
  for (int e = 0; e < 16; ++e) {
    int k = (e < 8) ? (g * 8 + e) : (16 + g * 8 + (e - 8));
    b[e] = __builtin_bit_cast(__bf16, p[n * sn + k * sk]);
  }
  return b;
}

// ---------------------------------------------------------------------------
// fp32 -> bf16 cast
// ---------------------------------------------------------------------------
__global__ __launch_bounds__(256) void hc_cast_bf16(const float* __restrict__ in,
                                                    unsigned short* __restrict__ out,
                                                    long long n) {
  long long i = (long long)blockIdx.x * blockDim.x + threadIdx.x;
  long long stride = (long long)gridDim.x * blockDim.x;
  for (; i < n; i += stride) out[i] = f2bf(in[i]);
}

// ---------------------------------------------------------------------------
// Generic batched GEMM: C[m,n] = alpha * sum_k A[m,k]*B[n,k] + bias[n]
// A: bf16, row stride aRow (col stride 1).
// B: bf16, general strides (bRow over n, bCol over k):
//   bCol==1 -> vector loads along k (weights, K-matrix)
//   bRow==1 -> vector loads along n (V^T operand for P@V)
//   else    -> scalar gather
// Batch z: outer = z/innerMod, inner = z%innerMod; per-matrix offsets.
// Output: f32 (Cf) or bf16 (Cb).
// Block: 256 threads (8 waves, 4x2), macro tile 128x128, K-step 32.
// ---------------------------------------------------------------------------
__global__ __launch_bounds__(256) void hc_gemm_bf16(
    const unsigned short* __restrict__ A, long long aOuter, long long aInner, long long aRow,
    const unsigned short* __restrict__ Bmat, long long bOuter, long long bInner,
    long long bRow, long long bCol,
    float* __restrict__ Cf, unsigned short* __restrict__ Cb,
    long long cOuter, long long cInner, long long cRow,
    const float* __restrict__ bias, int K, int innerMod, float alpha) {
  __shared__ unsigned short As[BM * PADK];
  __shared__ unsigned short Bs[BN * PADK];

  int z = blockIdx.z;
  int zo = z / innerMod, zi = z % innerMod;
  const unsigned short* Ab = A + (long long)zo * aOuter + (long long)zi * aInner;
  const unsigned short* Bb = Bmat + (long long)zo * bOuter + (long long)zi * bInner;
  long long coff = (long long)zo * cOuter + (long long)zi * cInner;

  int bm = blockIdx.y * BM;
  int bn = blockIdx.x * BN;
  int tid = threadIdx.x;
  int wid = tid >> 5;
  int wm = wid >> 1, wn = wid & 1;     // 4 x 2 wave grid; wave tile 32x64

  v8f acc[2][4] = {};

  for (int k0 = 0; k0 < K; k0 += BKc) {
    // --- stage A tile (128x32): 512 x 16B, 2 per thread ---
#pragma unroll
    for (int it = 0; it < 2; ++it) {
      int i = tid + it * 256;
      int r = i >> 2;
      int c8 = (i & 3) * 8;
      cp16(Ab + (long long)(bm + r) * aRow + (k0 + c8), &As[r * PADK + c8]);
    }
    // --- stage B tile (128x32) ---
    if (bCol == 1) {
#pragma unroll
      for (int it = 0; it < 2; ++it) {
        int i = tid + it * 256;
        int r = i >> 2;
        int c8 = (i & 3) * 8;
        cp16(Bb + (long long)(bn + r) * bRow + (k0 + c8), &Bs[r * PADK + c8]);
      }
    } else if (bRow == 1) {
      // contiguous along n: b128 global loads, scatter-transpose into LDS
#pragma unroll
      for (int it = 0; it < 2; ++it) {
        int i = tid + it * 256;       // 0..511
        int kk = i >> 4;              // 0..31
        int n8 = (i & 15) * 8;        // 0..120
        uint4 vv = *reinterpret_cast<const uint4*>(
            Bb + (long long)(k0 + kk) * bCol + (bn + n8));
        const unsigned short* pv = reinterpret_cast<const unsigned short*>(&vv);
#pragma unroll
        for (int e = 0; e < 8; ++e) Bs[(n8 + e) * PADK + kk] = pv[e];
      }
    } else {
#pragma unroll
      for (int it = 0; it < 16; ++it) {
        int i = tid + it * 256;       // 0..4095
        int r = i >> 5;
        int c = i & 31;
        Bs[r * PADK + c] = Bb[(long long)(bn + r) * bRow + (long long)(k0 + c) * bCol];
      }
    }
    if (k0 + BKc < K) {
      __builtin_prefetch((const void*)(Ab + (long long)(bm + (tid >> 2)) * aRow +
                                       (k0 + BKc)), 0, 1);
    }
    stage_fence();

    v16bf af[2], bf[4];
    af[0] = ldA16x32(&As[(wm * 32) * PADK], PADK, 1);
    af[1] = ldA16x32(&As[(wm * 32 + 16) * PADK], PADK, 1);
#pragma unroll
    for (int j = 0; j < 4; ++j)
      bf[j] = ldB32x16(&Bs[(wn * 64 + j * 16) * PADK], PADK, 1);
#pragma unroll
    for (int i = 0; i < 2; ++i)
#pragma unroll
      for (int j = 0; j < 4; ++j)
        acc[i][j] = wmma_bf16(af[i], bf[j], acc[i][j]);
    __syncthreads();
  }

  // --- epilogue: C/D layout -> lane (cg selects M half, cn = N) ---
  int lane = tid & 31;
  int cn = lane & 15, cg = lane >> 4;
#pragma unroll
  for (int i = 0; i < 2; ++i) {
#pragma unroll
    for (int j = 0; j < 4; ++j) {
      int rb = bm + wm * 32 + i * 16 + cg * 8;
      int col = bn + wn * 64 + j * 16 + cn;
      float bv = bias ? bias[col] : 0.f;
#pragma unroll
      for (int r = 0; r < 8; ++r) {
        float val = alpha * acc[i][j][r] + bv;
        long long idx = coff + (long long)(rb + r) * cRow + col;
        if (Cb) Cb[idx] = f2bf(val);
        else    Cf[idx] = val;
      }
    }
  }
}

// ---------------------------------------------------------------------------
// Local windowed attention: one wave per (b, window, head).
// scores(32x32) = Q(32x512) K^T via WMMA, mask+softmax in LDS, then P@V.
// qkv layout: [B*S, 3*D], q cols h*512.., k at +D, v at +2D.
// Output heads: [B*S, D] bf16 at col h*512.
// ---------------------------------------------------------------------------
__global__ __launch_bounds__(32) void hc_local_attn(
    const unsigned short* __restrict__ qkv, const unsigned char* __restrict__ mask,
    unsigned short* __restrict__ heads, float scale) {
  __shared__ float sS[32][33];
  __shared__ unsigned short sP[32][PADK];

  int z = blockIdx.x;
  int h = z & (Hc - 1);
  int w = (z >> 3) & (NWc - 1);
  int b = z >> 9;
  long long rowbase = (long long)b * Sc + (long long)w * Wc;
  const unsigned short* qp = qkv + rowbase * (3LL * Dc) + (long long)h * DHc;
  const unsigned short* kp = qp + Dc;
  const unsigned short* vp = qp + 2LL * Dc;
  const int ldq = 3 * Dc;

  v8f s00 = {}, s01 = {}, s10 = {}, s11 = {};
  for (int kc = 0; kc < DHc; kc += 32) {
    v16bf a0 = ldA16x32(qp + kc, ldq, 1);
    v16bf a1 = ldA16x32(qp + 16LL * ldq + kc, ldq, 1);
    v16bf b0 = ldB32x16(kp + kc, ldq, 1);
    v16bf b1 = ldB32x16(kp + 16LL * ldq + kc, ldq, 1);
    s00 = wmma_bf16(a0, b0, s00);
    s01 = wmma_bf16(a0, b1, s01);
    s10 = wmma_bf16(a1, b0, s10);
    s11 = wmma_bf16(a1, b1, s11);
  }

  int lane = threadIdx.x & 31;
  int cn = lane & 15, cg = lane >> 4;
  v8f ss[2][2] = {{s00, s01}, {s10, s11}};
#pragma unroll
  for (int i = 0; i < 2; ++i)
#pragma unroll
    for (int j = 0; j < 2; ++j)
#pragma unroll
      for (int r = 0; r < 8; ++r)
        sS[i * 16 + cg * 8 + r][j * 16 + cn] = ss[i][j][r];
  __syncthreads();

  // softmax: each lane owns one row
  {
    int vi = mask[(long long)b * Sc + w * Wc + lane] != 0;
    float vals[32];
    float mx = -3e38f;
#pragma unroll
    for (int j = 0; j < 32; ++j) {
      int vj = mask[(long long)b * Sc + w * Wc + j] != 0;
      float v = sS[lane][j] * scale + ((vi && vj) ? 0.f : -1e9f);
      vals[j] = v;
      mx = fmaxf(mx, v);
    }
    float sum = 0.f;
#pragma unroll
    for (int j = 0; j < 32; ++j) { vals[j] = __expf(vals[j] - mx); sum += vals[j]; }
    float inv = 1.f / sum;
#pragma unroll
    for (int j = 0; j < 32; ++j) sP[lane][j] = f2bf(vals[j] * inv);
  }
  __syncthreads();

  // out(32x512) = P(32x32) @ V(32x512)
  v16bf p0 = ldA16x32(&sP[0][0], PADK, 1);
  v16bf p1 = ldA16x32(&sP[16][0], PADK, 1);
  unsigned short* hp = heads + rowbase * Dc + (long long)h * DHc;
  for (int nt = 0; nt < DHc / 16; ++nt) {
    // B = V^T chunk: n (=dh) stride 1, k (=key row) stride ldq
    v16bf vb = ldB32x16(vp + nt * 16, 1, ldq);
    v8f z8 = {};
    v8f o0 = wmma_bf16(p0, vb, z8);
    v8f o1 = wmma_bf16(p1, vb, z8);
#pragma unroll
    for (int r = 0; r < 8; ++r) {
      hp[(long long)(cg * 8 + r) * Dc + nt * 16 + cn]        = f2bf(o0[r]);
      hp[(long long)(16 + cg * 8 + r) * Dc + nt * 16 + cn]   = f2bf(o1[r]);
    }
  }
}

// ---------------------------------------------------------------------------
// Row softmax with key-padding mask: rows = B*H*T, cols = S. Writes bf16 P.
// ---------------------------------------------------------------------------
__global__ __launch_bounds__(256) void hc_softmax_g(
    const float* __restrict__ sc, const unsigned char* __restrict__ mask,
    unsigned short* __restrict__ P) {
  int row = blockIdx.x;
  int b = row / (Hc * Tc);
  const float* src = sc + (long long)row * Sc;
  unsigned short* dst = P + (long long)row * Sc;
  __shared__ float red[256];
  int tid = threadIdx.x;
  float v[Sc / 256];
  float mx = -3e38f;
#pragma unroll
  for (int i = 0; i < Sc / 256; ++i) {
    int j = tid + i * 256;
    float s = src[j] + (mask[(long long)b * Sc + j] ? 0.f : -1e9f);
    v[i] = s;
    mx = fmaxf(mx, s);
  }
  red[tid] = mx; __syncthreads();
  for (int k = 128; k > 0; k >>= 1) {
    if (tid < k) red[tid] = fmaxf(red[tid], red[tid + k]);
    __syncthreads();
  }
  mx = red[0]; __syncthreads();
  float sum = 0.f;
#pragma unroll
  for (int i = 0; i < Sc / 256; ++i) { v[i] = __expf(v[i] - mx); sum += v[i]; }
  red[tid] = sum; __syncthreads();
  for (int k = 128; k > 0; k >>= 1) {
    if (tid < k) red[tid] += red[tid + k];
    __syncthreads();
  }
  float inv = 1.f / red[0];
#pragma unroll
  for (int i = 0; i < Sc / 256; ++i) dst[tid + i * 256] = f2bf(v[i] * inv);
}

// ---------------------------------------------------------------------------
// y = LayerNorm(x[row % xMod] + resid[row]) * g + b  (bf16 out). D = 4096.
// ---------------------------------------------------------------------------
__global__ __launch_bounds__(256) void hc_ln_resid(
    const float* __restrict__ x, long long xMod, const float* __restrict__ rsd,
    const float* __restrict__ g, const float* __restrict__ be,
    unsigned short* __restrict__ outb) {
  long long row = blockIdx.x;
  const float* xp = x + (row % xMod) * (long long)Dc;
  const float* rp = rsd + row * (long long)Dc;
  unsigned short* op = outb + row * (long long)Dc;
  __shared__ float red[256];
  int tid = threadIdx.x;
  float v[Dc / 256];
  float s = 0.f, ss = 0.f;
#pragma unroll
  for (int i = 0; i < Dc / 256; ++i) {
    float t = xp[tid + i * 256] + rp[tid + i * 256];
    v[i] = t; s += t; ss += t * t;
  }
  red[tid] = s; __syncthreads();
  for (int k = 128; k > 0; k >>= 1) { if (tid < k) red[tid] += red[tid + k]; __syncthreads(); }
  float mean = red[0] / Dc; __syncthreads();
  red[tid] = ss; __syncthreads();
  for (int k = 128; k > 0; k >>= 1) { if (tid < k) red[tid] += red[tid + k]; __syncthreads(); }
  float var = red[0] / Dc - mean * mean;
  float inv = rsqrtf(var + 1e-5f);
#pragma unroll
  for (int i = 0; i < Dc / 256; ++i) {
    int c = tid + i * 256;
    op[c] = f2bf((v[i] - mean) * inv * g[c] + be[c]);
  }
}

// ---------------------------------------------------------------------------
// Host-side orchestration
// ---------------------------------------------------------------------------
extern "C" void kernel_launch(void* const* d_in, const int* in_sizes, int n_in,
                              void* d_out, int out_size, void* d_ws, size_t ws_size,
                              hipStream_t stream) {
  (void)in_sizes; (void)n_in; (void)out_size; (void)ws_size;
  const float*         vlm  = (const float*)d_in[0];
  const unsigned char* mask = (const unsigned char*)d_in[1];
  const float* wli_f = (const float*)d_in[2];
  const float* bli   = (const float*)d_in[3];
  const float* wlo_f = (const float*)d_in[4];
  const float* blo   = (const float*)d_in[5];
  const float* lng   = (const float*)d_in[6];
  const float* lnbv  = (const float*)d_in[7];
  const float* wlp_f = (const float*)d_in[8];
  const float* blp   = (const float*)d_in[9];
  const float* wgi_f = (const float*)d_in[10];
  const float* bgi   = (const float*)d_in[11];
  const float* wgo_f = (const float*)d_in[12];
  const float* bgo   = (const float*)d_in[13];
  const float* gng   = (const float*)d_in[14];
  const float* gnb   = (const float*)d_in[15];
  const float* qt    = (const float*)d_in[16];
  const float* wop_f = (const float*)d_in[17];
  const float* bop   = (const float*)d_in[18];
  float* out = (float*)d_out;

  // bump allocator in workspace
  char* base = (char*)d_ws;
  size_t off = 0;
  auto alloc = [&](size_t bytes) -> void* {
    void* p = base + off;
    off = (off + bytes + 255) & ~(size_t)255;
    return p;
  };
  const long long BS = (long long)Bc * Sc;      // 8192
  const long long BT = (long long)Bc * Tc;      // 1024

  unsigned short* wli   = (unsigned short*)alloc(3LL * Dc * Dc * 2);
  unsigned short* wlo   = (unsigned short*)alloc((long long)Dc * Dc * 2);
  unsigned short* wlp   = (unsigned short*)alloc((long long)Dc * Dc * 2);
  unsigned short* wgi   = (unsigned short*)alloc(3LL * Dc * Dc * 2);
  unsigned short* wgo   = (unsigned short*)alloc((long long)Dc * Dc * 2);
  unsigned short* wop   = (unsigned short*)alloc((long long)Dc * Dc * 2);
  unsigned short* xbf   = (unsigned short*)alloc(BS * Dc * 2);
  unsigned short* qkvl  = (unsigned short*)alloc(BS * 3LL * Dc * 2);
  unsigned short* headl = (unsigned short*)alloc(BS * Dc * 2);
  float*          attnl = (float*)alloc(BS * Dc * 4);
  unsigned short* locbf = (unsigned short*)alloc(BS * Dc * 2);
  unsigned short* lfbf  = (unsigned short*)alloc(BS * Dc * 2);
  unsigned short* qtok  = (unsigned short*)alloc((long long)Tc * Dc * 2);
  unsigned short* qg    = (unsigned short*)alloc(BT * Dc * 2);
  unsigned short* kvg   = (unsigned short*)alloc(BS * 2LL * Dc * 2);
  float*          scrs  = (float*)alloc((long long)Bc * Hc * Tc * Sc * 4);
  unsigned short* Pbf   = (unsigned short*)alloc((long long)Bc * Hc * Tc * Sc * 2);
  unsigned short* headg = (unsigned short*)alloc(BT * Dc * 2);
  float*          attng = (float*)alloc(BT * Dc * 4);
  unsigned short* globf = (unsigned short*)alloc(BT * Dc * 2);

  auto cast = [&](const float* src, unsigned short* dst, long long n) {
    hc_cast_bf16<<<4096, 256, 0, stream>>>(src, dst, n);
  };
  auto gemm = [&](const unsigned short* A, long long aO, long long aI, long long aR,
                  const unsigned short* Bm, long long bO, long long bI, long long bR,
                  long long bC, float* Cf, unsigned short* Cb, long long cO,
                  long long cI, long long cR, const float* bias, int M, int N, int K,
                  int batch, int innerMod, float alpha) {
    dim3 grid(N / BN, M / BM, batch);
    hc_gemm_bf16<<<grid, 256, 0, stream>>>(A, aO, aI, aR, Bm, bO, bI, bR, bC, Cf, Cb,
                                           cO, cI, cR, bias, K, innerMod, alpha);
  };

  // ---- casts (weights + activations) ----
  cast(wli_f, wli, 3LL * Dc * Dc);
  cast(wlo_f, wlo, (long long)Dc * Dc);
  cast(wlp_f, wlp, (long long)Dc * Dc);
  cast(wgi_f, wgi, 3LL * Dc * Dc);
  cast(wgo_f, wgo, (long long)Dc * Dc);
  cast(wop_f, wop, (long long)Dc * Dc);
  cast(vlm, xbf, BS * Dc);
  cast(qt, qtok, (long long)Tc * Dc);

  const float scale = 1.0f / sqrtf((float)DHc);

  // ---- local path ----
  // qkv_l = x @ local_in_w^T + b : [8192, 12288] bf16
  gemm(xbf, 0, 0, Dc, wli, 0, 0, Dc, 1, nullptr, qkvl, 0, 0, 3LL * Dc, bli,
       (int)BS, 3 * Dc, Dc, 1, 1, 1.0f);
  // windowed attention -> headl [8192, 4096] bf16
  hc_local_attn<<<Bc * NWc * Hc, 32, 0, stream>>>(qkvl, mask, headl, scale);
  // attn_l = headl @ local_out_w^T + b : f32
  gemm(headl, 0, 0, Dc, wlo, 0, 0, Dc, 1, attnl, nullptr, 0, 0, Dc, blo,
       (int)BS, Dc, Dc, 1, 1, 1.0f);
  // loc = LN(x + attn_l) -> bf16
  hc_ln_resid<<<(int)BS, 256, 0, stream>>>(vlm, BS, attnl, lng, lnbv, locbf);
  // local_features = loc @ local_proj_w^T + b -> bf16
  gemm(locbf, 0, 0, Dc, wlp, 0, 0, Dc, 1, nullptr, lfbf, 0, 0, Dc, blp,
       (int)BS, Dc, Dc, 1, 1, 1.0f);

  // ---- global path ----
  // q_g = queries @ Wq^T + bq : batch over B with broadcast A
  gemm(qtok, 0, 0, Dc, wgi, 0, 0, Dc, 1, nullptr, qg, (long long)Tc * Dc, 0, Dc,
       bgi, Tc, Dc, Dc, Bc, 1, 1.0f);
  // kv_g = local_features @ [Wk;Wv]^T + b : [8192, 8192] bf16
  gemm(lfbf, 0, 0, Dc, wgi + (long long)Dc * Dc, 0, 0, Dc, 1, nullptr, kvg, 0, 0,
       2LL * Dc, bgi + Dc, (int)BS, 2 * Dc, Dc, 1, 1, 1.0f);
  // scores = scale * q @ k^T : batch B*H, f32 [B,H,T,S]
  gemm(qg, (long long)Tc * Dc, DHc, Dc,
       kvg, (long long)Sc * 2 * Dc, DHc, 2LL * Dc, 1,
       scrs, nullptr, (long long)Hc * Tc * Sc, (long long)Tc * Sc, Sc,
       nullptr, Tc, Sc, DHc, Bc * Hc, Hc, scale);
  // softmax over S with key-padding mask -> Pbf
  hc_softmax_g<<<Bc * Hc * Tc, 256, 0, stream>>>(scrs, mask, Pbf);
  // heads_g = P @ V : batch B*H, out bf16 [B,T,D]  (B contiguous along n)
  gemm(Pbf, (long long)Hc * Tc * Sc, (long long)Tc * Sc, Sc,
       kvg + Dc, (long long)Sc * 2 * Dc, DHc, 1, 2LL * Dc,
       nullptr, headg, (long long)Tc * Dc, DHc, Dc,
       nullptr, Tc, DHc, Sc, Bc * Hc, Hc, 1.0f);
  // attn_g = heads_g @ global_out_w^T + b : f32
  gemm(headg, 0, 0, Dc, wgo, 0, 0, Dc, 1, attng, nullptr, 0, 0, Dc, bgo,
       (int)BT, Dc, Dc, 1, 1, 1.0f);
  // glob = LN(queries + attn_g) -> bf16  (queries broadcast via row % T)
  hc_ln_resid<<<(int)BT, 256, 0, stream>>>(qt, (long long)Tc, attng, gng, gnb, globf);
  // compressed = glob @ output_proj_w^T + b : f32 -> d_out
  gemm(globf, 0, 0, Dc, wop, 0, 0, Dc, 1, out, nullptr, 0, 0, Dc, bop,
       (int)BT, Dc, Dc, 1, 1, 1.0f);
}
